// MambaSeqBlock_64785286693104
// MI455X (gfx1250) — compile-verified, matrix operand order checked by reading, not run on hardware
//
#include <hip/hip_runtime.h>

// ---------------------------------------------------------------------------
// Mamba2 block for MI455X (gfx1250, wave32, WMMA).
//
// Roofline: ~120 GFLOP of GEMM vs ~160MB of traffic -> compute bound on the
// matrix pipes. All four big GEMMs run through v_wmma_f32_16x16x32_f16 with
// fp16 operands / f32 accumulation. Weights are transposed to N x K fp16 so
// A- and B-fragment loads are identical contiguous b128 accesses per lane.
// Global->LDS staging uses CDNA5 async copies (global_load_async_to_lds_b128,
// ASYNCcnt) instead of VGPR round-trips.
// ---------------------------------------------------------------------------

#define D_MODEL   1024
#define D_STATE   16
#define NHEADS    32
#define HEADDIM   64
#define D_INNER   2048
#define CONV_DIM  2080      // D_INNER + 2*D_STATE
#define D_IN_PROJ 4160      // 2*D_INNER + 2*D_STATE + NHEADS
#define D_MLP     4096
#define BATCH     2
#define SEQLEN    2048
#define NTOK      (BATCH * SEQLEN)   // 4096

typedef _Float16 f16x8  __attribute__((ext_vector_type(8)));
typedef _Float16 f16x16 __attribute__((ext_vector_type(16)));
typedef float    f32x8  __attribute__((ext_vector_type(8)));

union V16U { f16x16 v; f16x8 h[2]; };

__device__ __forceinline__ float dev_silu(float v) {
    return v / (1.0f + expf(-v));
}

// CDNA5 async global->LDS copy (16B). The LDS operand VGPR holds the
// wave-relative LDS byte address; a generic LDS pointer's low 32 bits are
// exactly that (flat->LDS mapping truncates to addr[31:0], ISA 10.2).
__device__ __forceinline__ void async_copy_b128(const _Float16* gptr,
                                                const _Float16* lptr) {
    unsigned lds = (unsigned)(unsigned long long)lptr;
    asm volatile("global_load_async_to_lds_b128 %0, %1, off"
                 :: "v"(lds), "v"(gptr) : "memory");
}
__device__ __forceinline__ void async_wait0() {
    asm volatile("s_wait_asynccnt 0x0" ::: "memory");
}

// ---------------------------------------------------------------------------
// Transpose + cast fp32 (K x N) -> fp16 (N x K).  All dims are multiples of 32.
// ---------------------------------------------------------------------------
__global__ __launch_bounds__(256)
void k_transpose_cast(const float* __restrict__ src, _Float16* __restrict__ dst,
                      int K, int N) {
    __shared__ float tile[32][33];
    const int n0 = blockIdx.x * 32;
    const int k0 = blockIdx.y * 32;
    const int tx = threadIdx.x;   // 0..31
    const int ty = threadIdx.y;   // 0..7
#pragma unroll
    for (int j = 0; j < 32; j += 8)
        tile[ty + j][tx] = src[(size_t)(k0 + ty + j) * N + n0 + tx];
    __syncthreads();
#pragma unroll
    for (int j = 0; j < 32; j += 8)
        dst[(size_t)(n0 + ty + j) * K + k0 + tx] = (_Float16)tile[tx][ty + j];
}

// ---------------------------------------------------------------------------
// RMSNorm row kernel: out_f16[t, :] = x * rsqrt(mean(x^2)+eps) * w
// ---------------------------------------------------------------------------
__global__ __launch_bounds__(256)
void k_rmsnorm_f16(const float* __restrict__ x, const float* __restrict__ w,
                   _Float16* __restrict__ out, int D, float eps) {
    const int t   = blockIdx.x;
    const int tid = threadIdx.x;
    const int lane = tid & 31, wid = tid >> 5;
    const float* row = x + (size_t)t * D;

    float s = 0.f;
    for (int i = tid; i < D; i += 256) { float v = row[i]; s += v * v; }
#pragma unroll
    for (int o = 16; o; o >>= 1) s += __shfl_xor(s, o, 32);
    __shared__ float red[8];
    if (lane == 0) red[wid] = s;
    __syncthreads();
    if (tid == 0) {
        float tot = 0.f;
#pragma unroll
        for (int i = 0; i < 8; ++i) tot += red[i];
        red[0] = rsqrtf(tot / (float)D + eps);
    }
    __syncthreads();
    const float inv = red[0];
    _Float16* orow = out + (size_t)t * D;
    for (int i = tid; i < D; i += 256)
        orow[i] = (_Float16)(row[i] * inv * w[i]);
}

// ---------------------------------------------------------------------------
// Gated RMSNorm: v = y * silu(z);  out_f16 = v * rsqrt(mean(v^2)+eps) * gw
// ---------------------------------------------------------------------------
__global__ __launch_bounds__(256)
void k_gate_rmsnorm(const float* __restrict__ y, const float* __restrict__ zx,
                    const float* __restrict__ gw, _Float16* __restrict__ out) {
    const int t   = blockIdx.x;
    const int tid = threadIdx.x;
    const int lane = tid & 31, wid = tid >> 5;
    const float* yr = y  + (size_t)t * D_INNER;
    const float* zr = zx + (size_t)t * D_IN_PROJ;

    float s = 0.f;
    for (int i = tid; i < D_INNER; i += 256) {
        float v = yr[i] * dev_silu(zr[i]);
        s += v * v;
    }
#pragma unroll
    for (int o = 16; o; o >>= 1) s += __shfl_xor(s, o, 32);
    __shared__ float red[8];
    if (lane == 0) red[wid] = s;
    __syncthreads();
    if (tid == 0) {
        float tot = 0.f;
#pragma unroll
        for (int i = 0; i < 8; ++i) tot += red[i];
        red[0] = rsqrtf(tot / (float)D_INNER + 1e-5f);
    }
    __syncthreads();
    const float inv = red[0];
    _Float16* orow = out + (size_t)t * D_INNER;
    for (int i = tid; i < D_INNER; i += 256)
        orow[i] = (_Float16)(yr[i] * dev_silu(zr[i]) * inv * gw[i]);
}

// ---------------------------------------------------------------------------
// dt = softplus(raw + dt_bias),  dA = exp(dt * (-exp(A_log)))
// ---------------------------------------------------------------------------
__global__ __launch_bounds__(256)
void k_dtda(const float* __restrict__ zx, const float* __restrict__ dt_bias,
            const float* __restrict__ A_log, float* __restrict__ dt,
            float* __restrict__ dA) {
    const int idx = blockIdx.x * 256 + threadIdx.x;   // [0, NTOK*NHEADS)
    const int h = idx & (NHEADS - 1);
    const int t = idx >> 5;
    float v = zx[(size_t)t * D_IN_PROJ + (D_INNER + CONV_DIM) + h] + dt_bias[h];
    float sp = (v > 20.f) ? v : log1pf(expf(v));
    float A  = -expf(A_log[h]);
    dt[idx] = sp;
    dA[idx] = expf(sp * A);
}

// ---------------------------------------------------------------------------
// Causal depthwise conv (K=3) over xBC slice + silu.  One block per token.
// ---------------------------------------------------------------------------
__global__ __launch_bounds__(256)
void k_conv_silu(const float* __restrict__ zx, const float* __restrict__ cw,
                 const float* __restrict__ cb, float* __restrict__ out) {
    const int t = blockIdx.x;
    const int l = t & (SEQLEN - 1);
    const float* r0 = zx + (size_t)t * D_IN_PROJ + D_INNER;
    float* orow = out + (size_t)t * CONV_DIM;
    for (int c = threadIdx.x; c < CONV_DIM; c += 256) {
        float w0 = cw[c * 3 + 0], w1 = cw[c * 3 + 1], w2 = cw[c * 3 + 2];
        float xm2 = (l >= 2) ? r0[(ptrdiff_t)c - 2 * D_IN_PROJ] : 0.f;
        float xm1 = (l >= 1) ? r0[(ptrdiff_t)c - D_IN_PROJ]     : 0.f;
        float acc = cb[c] + w0 * xm2 + w1 * xm1 + w2 * r0[c];
        orow[c] = dev_silu(acc);
    }
}

// ---------------------------------------------------------------------------
// Selective state scan.  One block per (batch, head); thread = (p, n) with
// p in [0,64), n in [0,16).  State lives in a register; y reduction over n
// via 16-wide shuffle-xor within the wave32.
// ---------------------------------------------------------------------------
__global__ __launch_bounds__(1024)
void k_scan(const float* __restrict__ xbc, const float* __restrict__ dt,
            const float* __restrict__ dA, const float* __restrict__ Dv,
            float* __restrict__ y) {
    const int bh = blockIdx.x;
    const int b = bh >> 5;           // NHEADS == 32
    const int h = bh & 31;
    const int tid = threadIdx.x;
    const int p = tid >> 4;
    const int n = tid & 15;
    const float Dh = Dv[h];

    float state = 0.f;
    const size_t rbase  = (size_t)b * SEQLEN * CONV_DIM;
    const size_t dtbase = (size_t)b * SEQLEN * NHEADS + h;
    const size_t ybase  = (size_t)b * SEQLEN * D_INNER + h * HEADDIM;

    for (int l = 0; l < SEQLEN; ++l) {
        const float* row = xbc + rbase + (size_t)l * CONV_DIM;
        float xv  = row[h * HEADDIM + p];
        float Bv  = row[D_INNER + n];
        float Cv  = row[D_INNER + D_STATE + n];
        float dtv = dt[dtbase + (size_t)l * NHEADS];
        float dAv = dA[dtbase + (size_t)l * NHEADS];
        state = state * dAv + xv * Bv * dtv;
        float part = state * Cv;
#pragma unroll
        for (int o = 8; o; o >>= 1) part += __shfl_xor(part, o, 16);
        if (n == 0)
            y[ybase + (size_t)l * D_INNER + p] = part + Dh * xv;
    }
}

// ---------------------------------------------------------------------------
// WMMA GEMM:  C[M,N] = A[M,K] (fp16, row-major) * Bt[N,K] (fp16, "B^T").
// Block tile 256x64, 8 waves; wave w -> rows [w*32, w*32+32), 4 N-tiles
// -> 8 independent v_wmma per K-step (hides D->A/B hazards).
// K stepped by 32 through LDS (padded stride 40 halves, conflict-free b128
// fragment reads matching the 16-bit A 16x32 WMMA lane layout).
// Staging uses global_load_async_to_lds_b128 + s_wait_asynccnt.
// MODE: 0 = store f32; 1 = f32 + residual; 2 = bias+silu -> f16; 3 = bias+res.
// ---------------------------------------------------------------------------
#define BM 256
#define BN 64
#define BK 32
#define ASTR 40
#define BSTR 40

template <int MODE>
__global__ __launch_bounds__(256)
void k_gemm_wmma(const _Float16* __restrict__ A, const _Float16* __restrict__ Bt,
                 float* __restrict__ Cf, _Float16* __restrict__ Ch,
                 const float* __restrict__ bias, const float* __restrict__ res,
                 int M, int N, int K) {
    __shared__ _Float16 As[BM * ASTR];   // 20480 B
    __shared__ _Float16 Bs[BN * BSTR];   //  5120 B

    const int tid  = threadIdx.x;
    const int wave = tid >> 5;
    const int lane = tid & 31;
    const int m0 = blockIdx.x * BM;
    const int n0 = blockIdx.y * BN;
    const int mrow = lane & 15;
    const int kb   = (lane & 16) ? 8 : 0;   // K sub-offset for upper half-wave

    const f32x8 zero = {0.f, 0.f, 0.f, 0.f, 0.f, 0.f, 0.f, 0.f};
    f32x8 acc[2][4] = {{zero, zero, zero, zero}, {zero, zero, zero, zero}};

    for (int kk = 0; kk < K; kk += BK) {
        // --- async global -> LDS staging (b128 chunks, ASYNCcnt tracked) ---
#pragma unroll
        for (int it = 0; it < 4; ++it) {                 // A: 1024 chunks
            int c = tid + it * 256;
            int row = c >> 2, off = (c & 3) * 8;
            async_copy_b128(&A[(size_t)(m0 + row) * K + kk + off],
                            &As[row * ASTR + off]);
        }
        {                                                // B: 256 chunks
            int row = tid >> 2, off = (tid & 3) * 8;
            async_copy_b128(&Bt[(size_t)(n0 + row) * K + kk + off],
                            &Bs[row * BSTR + off]);
        }
        if (kk + BK < K) {                               // next-slab prefetch
            __builtin_prefetch(&A[(size_t)(m0 + tid) * K + kk + BK], 0, 1);
            __builtin_prefetch(&Bt[(size_t)(n0 + (tid & 63)) * K + kk + BK], 0, 1);
        }
        async_wait0();          // this wave's LDS writes landed
        __syncthreads();        // everyone's writes visible

        // --- fragments: lane m = lane&15; K = kb+0..7 and 16+kb..16+kb+7 ---
        V16U au[2];
#pragma unroll
        for (int mi = 0; mi < 2; ++mi) {
            const _Float16* ap = &As[(wave * 32 + mi * 16 + mrow) * ASTR + kb];
            au[mi].h[0] = *(const f16x8*)ap;
            au[mi].h[1] = *(const f16x8*)(ap + 16);
        }
#pragma unroll
        for (int t = 0; t < 4; ++t) {
            V16U bu;
            const _Float16* bp = &Bs[(t * 16 + mrow) * BSTR + kb];
            bu.h[0] = *(const f16x8*)bp;
            bu.h[1] = *(const f16x8*)(bp + 16);
#pragma unroll
            for (int mi = 0; mi < 2; ++mi)
                acc[mi][t] = __builtin_amdgcn_wmma_f32_16x16x32_f16(
                    false, au[mi].v, false, bu.v, (short)0, acc[mi][t],
                    false, false);
        }
        __syncthreads();        // protect LDS against next-iter restage
    }

    // --- epilogue: VGPR r of C/D holds M = r (+8 for lanes 16-31) ----------
#pragma unroll
    for (int mi = 0; mi < 2; ++mi) {
        const int crow0 = m0 + wave * 32 + mi * 16 + ((lane & 16) ? 8 : 0);
#pragma unroll
        for (int t = 0; t < 4; ++t) {
            const int col = n0 + t * 16 + mrow;
            const float bv = (MODE == 2 || MODE == 3) ? bias[col] : 0.f;
#pragma unroll
            for (int r = 0; r < 8; ++r) {
                const size_t idx = (size_t)(crow0 + r) * N + col;
                float v = acc[mi][t][r];
                if (MODE == 0) {
                    Cf[idx] = v;
                } else if (MODE == 1) {
                    Cf[idx] = v + res[idx];
                } else if (MODE == 2) {
                    v += bv;
                    Ch[idx] = (_Float16)dev_silu(v);
                } else {
                    Cf[idx] = v + bv + res[idx];
                }
            }
        }
    }
}

// ---------------------------------------------------------------------------
// Host-side orchestration
// ---------------------------------------------------------------------------
extern "C" void kernel_launch(void* const* d_in, const int* in_sizes, int n_in,
                              void* d_out, int out_size, void* d_ws, size_t ws_size,
                              hipStream_t stream) {
    (void)in_sizes; (void)n_in; (void)out_size; (void)ws_size;

    const float* x       = (const float*)d_in[0];
    const float* n1w     = (const float*)d_in[1];
    const float* W_in    = (const float*)d_in[2];
    const float* conv_w  = (const float*)d_in[3];
    const float* conv_b  = (const float*)d_in[4];
    const float* dt_bias = (const float*)d_in[5];
    const float* A_log   = (const float*)d_in[6];
    const float* Dvec    = (const float*)d_in[7];
    const float* ngw     = (const float*)d_in[8];
    const float* W_out   = (const float*)d_in[9];
    const float* n2w     = (const float*)d_in[10];
    const float* mlp_w1  = (const float*)d_in[11];
    const float* mlp_b1  = (const float*)d_in[12];
    const float* mlp_w2  = (const float*)d_in[13];
    const float* mlp_b2  = (const float*)d_in[14];
    float* out = (float*)d_out;

    // scratch carve-out (256B aligned), ~250 MB total
    char* wp = (char*)d_ws;
    auto alloc = [&](size_t bytes) -> void* {
        void* p = (void*)wp;
        wp += (bytes + 255) & ~(size_t)255;
        return p;
    };
    _Float16* h16    = (_Float16*)alloc((size_t)NTOK * D_MODEL * 2);
    _Float16* Wt_in  = (_Float16*)alloc((size_t)D_IN_PROJ * D_MODEL * 2);
    _Float16* Wt_out = (_Float16*)alloc((size_t)D_MODEL * D_INNER * 2);
    _Float16* Wt_m1  = (_Float16*)alloc((size_t)D_MLP * D_MODEL * 2);
    _Float16* Wt_m2  = (_Float16*)alloc((size_t)D_MODEL * D_MLP * 2);
    float*    zx     = (float*)alloc((size_t)NTOK * D_IN_PROJ * 4);
    float*    xbc    = (float*)alloc((size_t)NTOK * CONV_DIM * 4);
    float*    dtb    = (float*)alloc((size_t)NTOK * NHEADS * 4);
    float*    dab    = (float*)alloc((size_t)NTOK * NHEADS * 4);
    float*    ybuf   = (float*)alloc((size_t)NTOK * D_INNER * 4);
    _Float16* y16    = (_Float16*)alloc((size_t)NTOK * D_INNER * 2);
    float*    x1     = (float*)alloc((size_t)NTOK * D_MODEL * 4);
    _Float16* h2_16  = (_Float16*)alloc((size_t)NTOK * D_MODEL * 2);
    _Float16* mid16  = (_Float16*)alloc((size_t)NTOK * D_MLP * 2);

    const dim3 tb(32, 8);
    // weight transpose+cast (K x N fp32 -> N x K fp16)
    k_transpose_cast<<<dim3(D_IN_PROJ / 32, D_MODEL / 32), tb, 0, stream>>>(W_in,   Wt_in,  D_MODEL, D_IN_PROJ);
    k_transpose_cast<<<dim3(D_MODEL / 32,  D_INNER / 32), tb, 0, stream>>>(W_out,  Wt_out, D_INNER, D_MODEL);
    k_transpose_cast<<<dim3(D_MLP / 32,    D_MODEL / 32), tb, 0, stream>>>(mlp_w1, Wt_m1,  D_MODEL, D_MLP);
    k_transpose_cast<<<dim3(D_MODEL / 32,  D_MLP / 32),   tb, 0, stream>>>(mlp_w2, Wt_m2,  D_MLP,   D_MODEL);

    // h = rmsnorm(x, norm1_w) -> fp16
    k_rmsnorm_f16<<<NTOK, 256, 0, stream>>>(x, n1w, h16, D_MODEL, 1e-6f);

    // zxbcdt = h @ W_in
    k_gemm_wmma<0><<<dim3(NTOK / BM, D_IN_PROJ / BN), 256, 0, stream>>>(
        h16, Wt_in, zx, nullptr, nullptr, nullptr, NTOK, D_IN_PROJ, D_MODEL);

    // dt / dA, conv+silu, selective scan
    k_dtda<<<(NTOK * NHEADS) / 256, 256, 0, stream>>>(zx, dt_bias, A_log, dtb, dab);
    k_conv_silu<<<NTOK, 256, 0, stream>>>(zx, conv_w, conv_b, xbc);
    k_scan<<<BATCH * NHEADS, 1024, 0, stream>>>(xbc, dtb, dab, Dvec, ybuf);

    // gated RMSNorm -> fp16, then y @ W_out + x residual -> x1
    k_gate_rmsnorm<<<NTOK, 256, 0, stream>>>(ybuf, zx, ngw, y16);
    k_gemm_wmma<1><<<dim3(NTOK / BM, D_MODEL / BN), 256, 0, stream>>>(
        y16, Wt_out, x1, nullptr, nullptr, x, NTOK, D_MODEL, D_INNER);

    // MLP: h2 = rmsnorm(x1); mid = silu(h2@w1+b1); out = x1 + mid@w2 + b2
    k_rmsnorm_f16<<<NTOK, 256, 0, stream>>>(x1, n2w, h2_16, D_MODEL, 1e-6f);
    k_gemm_wmma<2><<<dim3(NTOK / BM, D_MLP / BN), 256, 0, stream>>>(
        h2_16, Wt_m1, nullptr, mid16, mlp_b1, nullptr, NTOK, D_MLP, D_MODEL);
    k_gemm_wmma<3><<<dim3(NTOK / BM, D_MODEL / BN), 256, 0, stream>>>(
        mid16, Wt_m2, out, nullptr, mlp_b2, x1, NTOK, D_MODEL, D_MLP);
}
